// Self_Attention_31688268709913
// MI455X (gfx1250) — compile-verified, matrix operand order checked by reading, not run on hardware
//
#include <hip/hip_runtime.h>
#include <hip/hip_bf16.h>
#include <cstdint>

// ---------------------------------------------------------------------------
// Types
// ---------------------------------------------------------------------------
typedef __bf16 bf16_t;
typedef __attribute__((ext_vector_type(16))) __bf16 v16bf;
typedef __attribute__((ext_vector_type(8)))  float  v8f;

union FragBF {
    v16bf        v;
    unsigned int u[8];
    uint4        q[2];
};

__device__ __forceinline__ unsigned int pack2bf(float a, float b) {
    unsigned short x = __builtin_bit_cast(unsigned short, (bf16_t)a);
    unsigned short y = __builtin_bit_cast(unsigned short, (bf16_t)b);
    return (unsigned int)x | ((unsigned int)y << 16);
}

__device__ __forceinline__ v8f wmma_bf16(v16bf a, v16bf b, v8f c) {
    return __builtin_amdgcn_wmma_f32_16x16x32_bf16(
        /*neg_a=*/false, a, /*neg_b=*/false, b,
        /*c_mod=*/(short)0, c, /*reuse_a=*/false, /*reuse_b=*/false);
}

#define LOG2E 1.4426950408889634f
#define SCL   (0.125f * LOG2E)     // fold 1/sqrt(64) into exp2 exponent

// Problem constants
#define BB 8
#define SS 2048
#define DD 512
#define HH 64

// ---------------------------------------------------------------------------
// Kernel 0: W[3][512][64] f32  ->  WT[3][64][512] bf16   (tiny, one-shot)
// ---------------------------------------------------------------------------
__global__ __launch_bounds__(256) void prep_wt(const float* __restrict__ W,
                                               bf16_t* __restrict__ WT) {
    int idx = blockIdx.x * 256 + threadIdx.x;      // 0 .. 3*512*64-1
    int mat = idx >> 15;                           // / (512*64)
    int r   = idx & 32767;
    int k   = r >> 6;
    int n   = r & 63;
    WT[((size_t)mat * HH + n) * DD + k] = (bf16_t)W[idx];
}

// ---------------------------------------------------------------------------
// Kernel 1: QKV projection.  One wave per (mat, b, 16-row tile).
//   Qd, Kd : [B][S][64] bf16        VT : [B][64][S] bf16 (V transposed)
// ---------------------------------------------------------------------------
__global__ __launch_bounds__(128) void qkv_proj(const float* __restrict__ x,
                                                const bf16_t* __restrict__ WT,
                                                bf16_t* __restrict__ Qd,
                                                bf16_t* __restrict__ Kd,
                                                bf16_t* __restrict__ VT) {
    const int lane = threadIdx.x & 31;
    const int wv   = threadIdx.x >> 5;
    const int tile = blockIdx.x * 4 + wv;          // 0 .. 3071
    const int mat  = tile >> 10;                   // / 1024
    const int rem  = tile & 1023;
    const int b    = rem >> 7;
    const int st   = rem & 127;
    const int q0   = st * 16;

    const int m    = lane & 15;                    // A-row / B-col / C-col
    const int koff = (lane >> 4) * 8;              // per-half K offset

    const float*  xrow  = x  + ((size_t)(b * SS + q0 + m)) * DD + koff;
    const bf16_t* wbase = WT + ((size_t)mat * HH) * DD;

    v8f acc[4];
    const v8f vz = {0.f, 0.f, 0.f, 0.f, 0.f, 0.f, 0.f, 0.f};
#pragma unroll
    for (int t = 0; t < 4; ++t) acc[t] = vz;

    for (int kk = 0; kk < DD / 32; ++kk) {
        // A fragment 16x32: convert f32 -> bf16 in-register
        const float4* pa0 = (const float4*)(xrow + kk * 32);
        const float4* pa1 = (const float4*)(xrow + kk * 32 + 16);
        float4 r0 = pa0[0], r1 = pa0[1];           // K = koff .. koff+7
        float4 r2 = pa1[0], r3 = pa1[1];           // K = 16+koff .. +7
        if (kk + 1 < DD / 32)                      // gfx1250 global_prefetch
            __builtin_prefetch(xrow + (kk + 1) * 32, 0, 1);
        FragBF fa;
        fa.u[0] = pack2bf(r0.x, r0.y); fa.u[1] = pack2bf(r0.z, r0.w);
        fa.u[2] = pack2bf(r1.x, r1.y); fa.u[3] = pack2bf(r1.z, r1.w);
        fa.u[4] = pack2bf(r2.x, r2.y); fa.u[5] = pack2bf(r2.z, r2.w);
        fa.u[6] = pack2bf(r3.x, r3.y); fa.u[7] = pack2bf(r3.z, r3.w);

        // Issue ALL B fragments first, then the WMMA quad (partial waits)
        FragBF fb[4];
#pragma unroll
        for (int t = 0; t < 4; ++t) {
            const bf16_t* wp = wbase + (size_t)(t * 16 + m) * DD + kk * 32 + koff;
            fb[t].q[0] = *(const uint4*)(wp);
            fb[t].q[1] = *(const uint4*)(wp + 16);
        }
#pragma unroll
        for (int t = 0; t < 4; ++t)
            acc[t] = wmma_bf16(fa.v, fb[t].v, acc[t]);
    }

    // Uniform store: one branch per wave, flat address arithmetic after
    const int rbase = (lane >> 4) * 8;             // C rows: v + rbase
    bf16_t* dst;
    size_t  rowStride, colStride;
    if (mat == 0) {
        dst = Qd + ((size_t)b * SS + q0 + rbase) * HH; rowStride = HH; colStride = 1;
    } else if (mat == 1) {
        dst = Kd + ((size_t)b * SS + q0 + rbase) * HH; rowStride = HH; colStride = 1;
    } else {
        dst = VT + (size_t)b * HH * SS + (q0 + rbase); rowStride = 1; colStride = SS;
    }
#pragma unroll
    for (int t = 0; t < 4; ++t) {
        const size_t ncol = (size_t)(t * 16 + m) * colStride;
#pragma unroll
        for (int v = 0; v < 8; ++v)
            dst[ncol + (size_t)v * rowStride] = (bf16_t)acc[t][v];
    }
}

// ---------------------------------------------------------------------------
// Async stage of one 32-key chunk of K (4KB contiguous) and V^T (64 x 64B
// rows) into LDS via gfx1250 global_load_async_to_lds_b128 (ASYNCcnt).
// Executed by one wave (all 32 lanes active): 16 async instructions, 8KB.
// ---------------------------------------------------------------------------
__device__ __forceinline__ void stage_chunk(int lane,
                                            const bf16_t* __restrict__ Kg,
                                            const bf16_t* __restrict__ Vg,
                                            unsigned kl, unsigned vl) {
    // K chunk: rows k0..k0+31 of [S][64] are contiguous -> flat 4KB copy
#pragma unroll
    for (int i = 0; i < 8; ++i) {
        unsigned off = (unsigned)((i * 32 + lane) * 16);
        const char* ga = (const char*)Kg + off;
        unsigned la = kl + off;
        asm volatile("global_load_async_to_lds_b128 %0, %1, off"
                     :: "v"(la), "v"(ga) : "memory");
    }
    // V chunk: 64 rows (h), 64B each, global row stride = S*2 bytes
#pragma unroll
    for (int i = 0; i < 8; ++i) {
        int s = i * 32 + lane;               // 0..255 16B-segments
        int n = s >> 2, c = s & 3;
        const char* ga = (const char*)(Vg + (size_t)n * SS) + c * 16;
        unsigned la = vl + (unsigned)(n * 64 + c * 16);
        asm volatile("global_load_async_to_lds_b128 %0, %1, off"
                     :: "v"(la), "v"(ga) : "memory");
    }
}

// ---------------------------------------------------------------------------
// Kernel 2: flash attention.  8 waves/block share one batch; K/V chunks are
// double-buffered in LDS, staged one iteration ahead by wave 0 (async DMA).
// ---------------------------------------------------------------------------
__global__ __launch_bounds__(256) void attn_kernel(const bf16_t* __restrict__ Qd,
                                                   const bf16_t* __restrict__ Kd,
                                                   const bf16_t* __restrict__ VT,
                                                   float* __restrict__ out) {
    __shared__ __align__(16) bf16_t Kbuf[2][32][64];    // 2 x 4KB
    __shared__ __align__(16) bf16_t Vbuf[2][64][32];    // 2 x 4KB
    __shared__ __align__(16) bf16_t Ptile[8][16][40];   // 80B row stride

    const int lane = threadIdx.x & 31;
    const int wv   = threadIdx.x >> 5;
    const int tile = blockIdx.x * 8 + wv;          // 0 .. 1023
    const int b    = tile >> 7;                    // uniform across block
    const int st   = tile & 127;
    const int q0   = st * 16;

    const int m    = lane & 15;
    const int hi   = lane >> 4;
    const int koff = hi * 8;

    const unsigned kbase = (unsigned)(uintptr_t)(&Kbuf[0][0][0]);
    const unsigned vbase = (unsigned)(uintptr_t)(&Vbuf[0][0][0]);
    const bf16_t* Kg0 = Kd + (size_t)b * SS * HH;       // chunk kc at +kc*32*64
    const bf16_t* Vg0 = VT + (size_t)b * HH * SS;       // chunk kc at +kc*32

    // Q fragments: 16x64 as two 16x32 bf16 A-fragments
    FragBF aq[2];
    {
        const bf16_t* qp = Qd + ((size_t)(b * SS + q0 + m)) * HH;
#pragma unroll
        for (int kk = 0; kk < 2; ++kk) {
            aq[kk].q[0] = *(const uint4*)(qp + kk * 32 + koff);
            aq[kk].q[1] = *(const uint4*)(qp + kk * 32 + 16 + koff);
        }
    }

    const v8f vz = {0.f, 0.f, 0.f, 0.f, 0.f, 0.f, 0.f, 0.f};
    float mrow[8], lsum[8];
#pragma unroll
    for (int v = 0; v < 8; ++v) { mrow[v] = -1e30f; lsum[v] = 0.f; }
    v8f o[4];
#pragma unroll
    for (int t = 0; t < 4; ++t) o[t] = vz;

    // Prologue: stage chunk 0 into buffer 0
    if (wv == 0) {
        stage_chunk(lane, Kg0, Vg0, kbase, vbase);
        asm volatile("s_wait_asynccnt 0" ::: "memory");
    }
    __syncthreads();

    for (int kc = 0; kc < SS / 32; ++kc) {
        const int cur = kc & 1, nxt = cur ^ 1;

        // Stage next chunk (async) while computing the current one
        if (wv == 0 && kc + 1 < SS / 32)
            stage_chunk(lane, Kg0 + (size_t)(kc + 1) * 32 * HH,
                        Vg0 + (size_t)(kc + 1) * 32,
                        kbase + (unsigned)nxt * 4096,
                        vbase + (unsigned)nxt * 4096);

        // ---- scores: 16x32 tile = two 16x16 C tiles, f32 accumulate ----
        v8f c[2];
#pragma unroll
        for (int g = 0; g < 2; ++g) {
            const int r = g * 16 + m;              // key index in chunk
            FragBF fb0, fb1;
            fb0.q[0] = *(const uint4*)(&Kbuf[cur][r][koff]);
            fb0.q[1] = *(const uint4*)(&Kbuf[cur][r][16 + koff]);
            fb1.q[0] = *(const uint4*)(&Kbuf[cur][r][32 + koff]);
            fb1.q[1] = *(const uint4*)(&Kbuf[cur][r][48 + koff]);
            v8f cz = vz;
            cz = wmma_bf16(aq[0].v, fb0.v, cz);
            cz = wmma_bf16(aq[1].v, fb1.v, cz);
            c[g] = cz;
        }

        // ---- online softmax over the 32 new keys (raw-score units) ----
        float p0[8], p1[8], rm[8], rs[8], scv[8];
#pragma unroll
        for (int v = 0; v < 8; ++v) {
            p0[v] = c[0][v];
            p1[v] = c[1][v];
            rm[v] = fmaxf(p0[v], p1[v]);
        }
#pragma unroll
        for (int v = 0; v < 8; ++v) {
#pragma unroll
            for (int msk = 1; msk < 16; msk <<= 1)
                rm[v] = fmaxf(rm[v], __shfl_xor(rm[v], msk, 32));
        }
#pragma unroll
        for (int v = 0; v < 8; ++v) {
            float mn = fmaxf(mrow[v], rm[v]);
            scv[v]   = exp2f((mrow[v] - mn) * SCL);
            mrow[v]  = mn;
            p0[v]    = exp2f((p0[v] - mn) * SCL);
            p1[v]    = exp2f((p1[v] - mn) * SCL);
            rs[v]    = p0[v] + p1[v];
        }
#pragma unroll
        for (int v = 0; v < 8; ++v) {
#pragma unroll
            for (int msk = 1; msk < 16; msk <<= 1)
                rs[v] += __shfl_xor(rs[v], msk, 32);
            lsum[v] = lsum[v] * scv[v] + rs[v];
        }
#pragma unroll
        for (int t = 0; t < 4; ++t)
#pragma unroll
            for (int v = 0; v < 8; ++v) o[t][v] *= scv[v];

        // ---- C-layout -> A-layout reswizzle of P through LDS ----
#pragma unroll
        for (int v = 0; v < 8; ++v) {
            Ptile[wv][hi * 8 + v][m]      = (bf16_t)p0[v];
            Ptile[wv][hi * 8 + v][16 + m] = (bf16_t)p1[v];
        }
        asm volatile("s_wait_dscnt 0" ::: "memory");   // wave-local RAW on LDS
        FragBF ap;
        ap.q[0] = *(const uint4*)(&Ptile[wv][m][koff]);
        ap.q[1] = *(const uint4*)(&Ptile[wv][m][16 + koff]);

        // ---- O += P @ V from LDS (V^T rows -> contiguous runs) ----
#pragma unroll
        for (int t = 0; t < 4; ++t) {
            const int n = t * 16 + m;
            FragBF fv;
            fv.q[0] = *(const uint4*)(&Vbuf[cur][n][koff]);
            fv.q[1] = *(const uint4*)(&Vbuf[cur][n][16 + koff]);
            o[t] = wmma_bf16(ap.v, fv.v, o[t]);
        }

        // Ensure next buffers are resident, then block-wide handoff
        if (wv == 0)
            asm volatile("s_wait_asynccnt 0" ::: "memory");
        __syncthreads();
    }

    // ---- epilogue: normalize and store f32 ----
#pragma unroll
    for (int v = 0; v < 8; ++v) lsum[v] = 1.0f / lsum[v];
#pragma unroll
    for (int t = 0; t < 4; ++t) {
        const int n = t * 16 + m;
#pragma unroll
        for (int v = 0; v < 8; ++v) {
            const int row = q0 + hi * 8 + v;
            out[((size_t)b * SS + row) * HH + n] = o[t][v] * lsum[v];
        }
    }
}

// ---------------------------------------------------------------------------
// Launcher
// ---------------------------------------------------------------------------
extern "C" void kernel_launch(void* const* d_in, const int* in_sizes, int n_in,
                              void* d_out, int out_size, void* d_ws, size_t ws_size,
                              hipStream_t stream) {
    const float* x = (const float*)d_in[0];   // [8][2048][512]
    const float* W = (const float*)d_in[1];   // [3][512][64]
    float* out = (float*)d_out;               // [8][2048][64]

    char* wsb = (char*)d_ws;
    const size_t WT_BYTES = (size_t)3 * HH * DD * sizeof(bf16_t);   // 196608
    const size_t QK_BYTES = (size_t)BB * SS * HH * sizeof(bf16_t);  // 2 MiB
    bf16_t* WT = (bf16_t*)(wsb);
    bf16_t* Qd = (bf16_t*)(wsb + WT_BYTES);
    bf16_t* Kd = (bf16_t*)(wsb + WT_BYTES + QK_BYTES);
    bf16_t* VT = (bf16_t*)(wsb + WT_BYTES + 2 * QK_BYTES);

    // W -> WT (bf16, transposed)
    prep_wt<<<(3 * DD * HH) / 256, 256, 0, stream>>>(W, WT);
    // QKV projection: 3 * 8 * 128 tiles, 4 waves / block
    qkv_proj<<<768, 128, 0, stream>>>(x, WT, Qd, Kd, VT);
    // flash attention: 1024 query tiles, 8 waves / block
    attn_kernel<<<128, 256, 0, stream>>>(Qd, Kd, VT, out);
}
